// FeedForwardQuantum_65481071400550
// MI455X (gfx1250) — matrix operand order repaired and checked by reading, not compiled, not run on hardware
//
#include <hip/hip_runtime.h>
#include <math.h>

typedef __attribute__((ext_vector_type(16))) _Float16 v16h;
typedef __attribute__((ext_vector_type(8)))  float    v8f;
typedef __attribute__((ext_vector_type(2)))  float    v2f;

#define TOKENS 8192
#define EMBED  1024
#define FFN    4096
#define NW     8
#define BM     128
#define BN     128
#define BK     32
#define NCHUNK (FFN / BK)   // 128
#define NTILES (EMBED / BN) // 8

// ---------------------------------------------------------------------------
// Prep: swizzle w2 (f32, [FFN, EMBED] row-major) into f16 B-fragment order:
//   [kc][ntile][slot(frag*32+lane)][j]  with  K = kc*32 + 8*(lane>=16) + j + (j&8),
//                                             N = ntile*128 + frag*16 + lane%16
// One 32-byte contiguous run per (chunk, ntile, slot).
// ---------------------------------------------------------------------------
__global__ __launch_bounds__(256)
void w2_swizzle(const float* __restrict__ w2, _Float16* __restrict__ w2s)
{
    const int gid = blockIdx.x * 256 + threadIdx.x;   // slot id: NCHUNK*NTILES*256
    const int s   = gid & 255;
    const int nt  = (gid >> 8) & (NTILES - 1);
    const int kc  = gid >> 11;
    const int fb  = s >> 5;
    const int l   = s & 31;
    const int N     = nt * BN + fb * 16 + (l & 15);
    const int Kbase = kc * BK + ((l >> 4) & 1) * 8;

    v16h hv;
    #pragma unroll
    for (int j = 0; j < 16; ++j)
        hv[j] = (_Float16)w2[(size_t)(Kbase + j + (j & 8)) * EMBED + N];
    *(v16h*)&w2s[(size_t)gid * 16] = hv;
}

// ---------------------------------------------------------------------------
// Fused: out = relu(cos(x[:, :8]+theta) @ w1) @ w2
// Block: 256 threads (8 waves). Output tile BM x BN, K-loop over FFN by BK.
// ---------------------------------------------------------------------------
template <bool PRE>
__global__ __launch_bounds__(256)
void ffq_fused(const float* __restrict__ x, const float* __restrict__ theta,
               const float* __restrict__ w1, const float* __restrict__ w2,
               const _Float16* __restrict__ w2s, float* __restrict__ out)
{
    __shared__ __align__(16) float    qs[BM * NW];          // 4 KB
    __shared__ __align__(16) _Float16 Ab[2][BM * BK];       // 2 x 8 KB, A-frag layout
    __shared__ __align__(16) _Float16 Bb[2][BK * BN];       // 2 x 8 KB, B-frag layout

    const int t    = threadIdx.x;
    const int lane = t & 31;
    const int wave = t >> 5;
    const int nBase = blockIdx.x * BN;
    const int mBase = blockIdx.y * BM;

    // ---------------- q = cos(x[:, :NW] + theta), once ----------------
    if (t < BM) {
        const float* xp = x + (size_t)(mBase + t) * EMBED;
        #pragma unroll
        for (int w = 0; w < NW; ++w)
            qs[t * NW + w] = cosf(xp[w] + theta[w]);
    }
    __syncthreads();

    // q^T fragments (B operand of the transposed h-GEMM), chunk-invariant.
    // B 4x16 f32 layout: lane l: N(token)=l%16, vgpr v -> K(wire)=2*(l>=16)+v.
    const int rg = wave;                    // row group handled by this wave in phase 1
    const int kq = ((lane >> 4) & 1) * 2;
    v2f q01, q23;
    {
        const int m16 = rg * 16 + (lane & 15);
        q01.x = qs[m16 * NW + kq];
        q01.y = qs[m16 * NW + kq + 1];
        q23.x = qs[m16 * NW + 4 + kq];
        q23.y = qs[m16 * NW + 4 + kq + 1];
    }

    // GEMM2 wave tiling: 2x4 waves, each wave 64 rows x 32 cols = 4x2 16x16 tiles.
    const int wm = wave >> 2;
    const int wn = wave & 3;

    v8f acc[4][2];
    #pragma unroll
    for (int mi = 0; mi < 4; ++mi)
        #pragma unroll
        for (int ni = 0; ni < 2; ++ni)
            acc[mi][ni] = (v8f){0.f, 0.f, 0.f, 0.f, 0.f, 0.f, 0.f, 0.f};

    // ------------- producer: h^T via f32 WMMA + w2 B-frags ----------------
    auto produce = [&](int kc, int bi) {
        const int kcBase = kc * BK;
        const int n_loc  = lane & 15;
        const int kb     = ((lane >> 4) & 1) * 2;   // A 16x4 f32 layout (wires)

        // h^T tile:  D[g] = w1_chunk^T (16 K-cols x 8 wires) x q^T (8 wires x 16 tokens)
        v8f hg[2];
        #pragma unroll
        for (int g = 0; g < 2; ++g) {
            const int col = kcBase + g * 16 + n_loc;   // w1 column == h K index
            v2f wa0, wa1;
            wa0.x = w1[(size_t)(kb + 0) * FFN + col];
            wa0.y = w1[(size_t)(kb + 1) * FFN + col];
            wa1.x = w1[(size_t)(kb + 4) * FFN + col];
            wa1.y = w1[(size_t)(kb + 5) * FFN + col];
            v8f hacc = (v8f){0.f, 0.f, 0.f, 0.f, 0.f, 0.f, 0.f, 0.f};
            hacc = __builtin_amdgcn_wmma_f32_16x16x4_f32(false, wa0, false, q01,
                                                         (short)0, hacc, false, false);
            hacc = __builtin_amdgcn_wmma_f32_16x16x4_f32(false, wa1, false, q23,
                                                         (short)0, hacc, false, false);
            hg[g] = hacc;
        }
        // D^T layout == A-fragment slot (rg, lane), element j = r + 8*g (table k order: k = 16g + 8h + r).
        // Convert first (v_cvt_pk_f16_f32), then relu in packed f16
        // (v_pk_max_num_f16) -- cvt output is canonical, so no quieting max pair.
        v16h ah;
        #pragma unroll
        for (int g = 0; g < 2; ++g)
            #pragma unroll
            for (int r = 0; r < 8; ++r)
                ah[r + 8 * g] = (_Float16)hg[g][r];
        const v16h z16 = {};
        ah = __builtin_elementwise_max(ah, z16);
        *(v16h*)&Ab[bi][(rg * 32 + lane) * 16] = ah;

        // w2 chunk -> B fragments (slot == thread id)
        if (PRE) {
            const _Float16* src =
                w2s + (((size_t)kc * NTILES + blockIdx.x) * 256 + t) * 16;
            unsigned int lds = (unsigned int)(uintptr_t)(void*)&Bb[bi][t * 16];
            unsigned long long ga = (unsigned long long)(uintptr_t)src;
            asm volatile("global_load_async_to_lds_b128 %0, %1, off\n\t"
                         "global_load_async_to_lds_b128 %0, %1, off offset:16"
                         :: "v"(lds), "v"(ga) : "memory");
        } else {
            const int khi = (lane >> 4) * 8;
            const float* w2p = w2 + (size_t)(kcBase + khi) * EMBED
                                  + nBase + wave * 16 + n_loc;
            v16h hv;
            #pragma unroll
            for (int j = 0; j < 16; ++j)
                hv[j] = (_Float16)w2p[(size_t)(j + (j & 8)) * EMBED];
            *(v16h*)&Bb[bi][t * 16] = hv;
        }
    };

    produce(0, 0);

    for (int kc = 0; kc < NCHUNK; ++kc) {
        if (PRE) asm volatile("s_wait_asynccnt 0x0" ::: "memory");
        __syncthreads();
        if (kc + 1 < NCHUNK) produce(kc + 1, (kc + 1) & 1);

        // ------------- consumer: 8 x v_wmma_f32_16x16x32_f16 per wave ------
        const int bi = kc & 1;
        v16h bf0 = *(const v16h*)&Bb[bi][((wn * 2 + 0) * 32 + lane) * 16];
        v16h bf1 = *(const v16h*)&Bb[bi][((wn * 2 + 1) * 32 + lane) * 16];
        #pragma unroll
        for (int mi = 0; mi < 4; ++mi) {
            v16h af = *(const v16h*)&Ab[bi][((wm * 4 + mi) * 32 + lane) * 16];
            acc[mi][0] = __builtin_amdgcn_wmma_f32_16x16x32_f16(
                false, af, false, bf0, (short)0, acc[mi][0], false, false);
            acc[mi][1] = __builtin_amdgcn_wmma_f32_16x16x32_f16(
                false, af, false, bf1, (short)0, acc[mi][1], false, false);
        }
    }

    // ---------------- epilogue: D layout -> global f32 ----------------
    const int hi8 = (lane >> 4) * 8;
    #pragma unroll
    for (int mi = 0; mi < 4; ++mi) {
        #pragma unroll
        for (int ni = 0; ni < 2; ++ni) {
            const int row0 = mBase + wm * 64 + mi * 16 + hi8;
            const int col  = nBase + wn * 32 + ni * 16 + (lane & 15);
            #pragma unroll
            for (int r = 0; r < 8; ++r)
                out[(size_t)(row0 + r) * EMBED + col] = acc[mi][ni][r];
        }
    }
}

extern "C" void kernel_launch(void* const* d_in, const int* in_sizes, int n_in,
                              void* d_out, int out_size, void* d_ws, size_t ws_size,
                              hipStream_t stream) {
    const float* x     = (const float*)d_in[0];
    const float* theta = (const float*)d_in[1];
    const float* w1    = (const float*)d_in[2];
    const float* w2    = (const float*)d_in[3];
    float* out = (float*)d_out;

    const size_t w2s_bytes = (size_t)FFN * EMBED * sizeof(_Float16);   // 8 MB
    dim3 grid(NTILES, TOKENS / BM);   // (8, 64) = 512 workgroups

    if (ws_size >= w2s_bytes) {
        _Float16* w2s = (_Float16*)d_ws;
        w2_swizzle<<<(NCHUNK * NTILES * 256) / 256, 256, 0, stream>>>(w2, w2s);
        ffq_fused<true><<<grid, 256, 0, stream>>>(x, theta, w1, w2, w2s, out);
    } else {
        ffq_fused<false><<<grid, 256, 0, stream>>>(x, theta, w1, w2, nullptr, out);
    }
}
